// HOPEBlock_32280974196820
// MI455X (gfx1250) — matrix operand order repaired and checked by reading, not compiled, not run on hardware
//
#include <hip/hip_runtime.h>
#include <hip/hip_bf16.h>
#include <math.h>

typedef __bf16 bf16_t;
typedef __attribute__((ext_vector_type(16))) __bf16 v16bf_t;
typedef __attribute__((ext_vector_type(8)))  __bf16 v8bf_t;
typedef __attribute__((ext_vector_type(8)))  float  v8f_t;

union FragAB { v16bf_t v; v8bf_t h[2]; };

// ---- problem constants (from setup_inputs: B=8, T=8, H=16, W=16) ----
constexpr int kBatch = 8;
constexpr int kN     = 8 * 16 * 16;      // 2048 tokens per batch entry
constexpr int kM     = kBatch * kN;      // 16384 total rows
constexpr int kD     = 384;
constexpr int kHid   = 1536;

// =====================================================================
// fp32 -> bf16 weight conversion WITH transpose:
// in:  K x N row-major (fp32)   out: N x K row-major (bf16) == W^T
// so GEMM B-tiles have contiguous K, identical layout to A-tiles.
// =====================================================================
__global__ void cvt_tr_kernel(const float* __restrict__ in,
                              bf16_t* __restrict__ out, int K, int N)
{
    int i = blockIdx.x * blockDim.x + threadIdx.x;   // over N*K
    if (i >= K * N) return;
    int n = i / K;
    int k = i - n * K;
    out[i] = (bf16_t)in[(size_t)k * N + n];
}

// =====================================================================
// fused (optional residual-add) + LayerNorm -> bf16
// one wave32 per row of 384; 12 elements per lane
// =====================================================================
__global__ __launch_bounds__(256) void resid_ln_kernel(
    const float* __restrict__ x, const float* __restrict__ delta,
    const float* __restrict__ scale, const float* __restrict__ bias,
    float* __restrict__ xout, bf16_t* __restrict__ ybf)
{
    int wave = threadIdx.x >> 5;
    int lane = threadIdx.x & 31;
    int row  = blockIdx.x * 8 + wave;
    size_t base = (size_t)row * kD;

    float v[12];
    float s = 0.f;
    #pragma unroll
    for (int j = 0; j < 12; ++j) {
        int c = lane + 32 * j;
        float t = x[base + c];
        if (delta) t += delta[base + c];
        v[j] = t;
        s += t;
    }
    #pragma unroll
    for (int m = 16; m > 0; m >>= 1) s += __shfl_xor(s, m, 32);
    float mean = s * (1.f / kD);

    float vs = 0.f;
    #pragma unroll
    for (int j = 0; j < 12; ++j) { float d = v[j] - mean; vs += d * d; }
    #pragma unroll
    for (int m = 16; m > 0; m >>= 1) vs += __shfl_xor(vs, m, 32);
    float rstd = rsqrtf(vs * (1.f / kD) + 1e-5f);

    #pragma unroll
    for (int j = 0; j < 12; ++j) {
        int c = lane + 32 * j;
        if (xout) xout[base + c] = v[j];
        float y = (v[j] - mean) * rstd * scale[c] + bias[c];
        ybf[base + c] = (bf16_t)y;
    }
}

// =====================================================================
// RoPE-3D on q (fp32 in, bf16 out). Head dim 24 = 3 segments of RD=8.
// =====================================================================
__global__ __launch_bounds__(256) void rope_kernel(
    const float* __restrict__ q, bf16_t* __restrict__ out)
{
    int g   = blockIdx.x * blockDim.x + threadIdx.x;   // global pair id
    int row = g / 192;                                 // 384/2 pairs per row
    int p   = g - row * 192;
    int c   = p * 2;
    int hc  = c % 24;          // offset within head
    int seg = hc >> 3;         // 0 = t, 1 = h, 2 = w
    int fi  = (hc & 7) >> 1;   // frequency index 0..3
    int n   = row & (kN - 1);  // token index within the sequence

    float pos = (seg == 0) ? (float)(n >> 8)
              : (seg == 1) ? (float)((n >> 4) & 15)
                           : (float)(n & 15);
    float invf = __expf(-2.3025851f * (float)fi);      // 10000^(-2*fi/8)
    float ang  = pos * invf;
    float sn, cs;
    __sincosf(ang, &sn, &cs);

    size_t idx = (size_t)row * kD + c;
    float x1 = q[idx], x2 = q[idx + 1];
    out[idx]     = (bf16_t)(x1 * cs - x2 * sn);
    out[idx + 1] = (bf16_t)(x1 * sn + x2 * cs);
}

// =====================================================================
// WMMA bf16 GEMM: C[M x N] = epilogue( A[M x K] @ W + bias ),
// with W passed PRE-TRANSPOSED (Wt is N x K row-major).
// 128x128 tile / block of 8 waves; wave = 32(M) x 64(N) -> 8 v8f accums.
// A and B tiles staged to double-buffered LDS with
// GLOBAL_LOAD_ASYNC_TO_LDS_B128 (ASYNCcnt), synced by s_wait_asynccnt.
// =====================================================================
enum { EPI_F32 = 0, EPI_BF16 = 1, EPI_GELU_BF16 = 2, EPI_ADD_F32 = 3 };

#define LDS_STRIDE 40   // 32 bf16 + 8 pad; 80B row stride keeps 16B alignment

template <int EPI>
__global__ __launch_bounds__(256) void gemm_bf16_kernel(
    const bf16_t* __restrict__ A, const bf16_t* __restrict__ Wt,
    const float* __restrict__ bias, void* __restrict__ Cv,
    int K, int N)
{
    // [buffer][row][k] ; row = M-row for A, N-col for B (Wt row)
    __shared__ __align__(16) bf16_t sA[2][128][LDS_STRIDE];
    __shared__ __align__(16) bf16_t sB[2][128][LDS_STRIDE];

    const int tid  = threadIdx.x;
    const int wave = tid >> 5;
    const int lane = tid & 31;
    const int hl   = lane >> 4;      // half-wave select
    const int lm   = lane & 15;
    const int row0 = blockIdx.x * 128;
    const int col0 = blockIdx.y * 128;
    const int wm   = (wave & 3) * 32;    // wave M offset in tile
    const int wn   = (wave >> 2) * 64;   // wave N offset in tile

    // LDS byte addresses (low 32 bits of the flat shared-aperture address
    // are the LDS offset on gfx1250)
    const uint32_t sAaddr = (uint32_t)(uintptr_t)&sA[0][0][0];
    const uint32_t sBaddr = (uint32_t)(uintptr_t)&sB[0][0][0];

    // staging: tile is 128 rows x 32 bf16 = 512 pieces of 16B; 2 per thread
    // piece q: row = q>>2, 8-elem segment = (q&3)*8
    auto stage = [&](int b, int k0) {
        #pragma unroll
        for (int p = 0; p < 2; ++p) {
            int q = p * 256 + tid;
            int r = q >> 2;
            int s = (q & 3) * 8;
            uint32_t la = sAaddr +
                (uint32_t)((b * 128 + r) * LDS_STRIDE + s) * 2u;
            uint32_t ga = (uint32_t)(((size_t)(row0 + r) * K + k0 + s) * 2u);
            asm volatile("global_load_async_to_lds_b128 %0, %1, %2"
                         :: "v"(la), "v"(ga), "s"(A) : "memory");
            uint32_t lb = sBaddr +
                (uint32_t)((b * 128 + r) * LDS_STRIDE + s) * 2u;
            uint32_t gb = (uint32_t)(((size_t)(col0 + r) * K + k0 + s) * 2u);
            asm volatile("global_load_async_to_lds_b128 %0, %1, %2"
                         :: "v"(lb), "v"(gb), "s"(Wt) : "memory");
        }
    };

    v8f_t acc[2][4];
    #pragma unroll
    for (int i = 0; i < 2; ++i)
        #pragma unroll
        for (int t = 0; t < 4; ++t)
            #pragma unroll
            for (int e = 0; e < 8; ++e) acc[i][t][e] = 0.f;

    const int nk = K / 32;
    stage(0, 0);

    for (int it = 0; it < nk; ++it) {
        // all of this wave's async fills done; barrier => whole tile ready and
        // everyone finished reading the buffer we are about to refill
        asm volatile("s_wait_asynccnt 0x0" ::: "memory");
        __syncthreads();
        if (it + 1 < nk) stage((it + 1) & 1, (it + 1) * 32);

        const int b = it & 1;
        // A fragment (16x32 bf16): lanes 0-15 row=lm K{0-7,16-23};
        // lanes 16-31 row=lm K{8-15,24-31}
        FragAB fa[2], fb[4];
        #pragma unroll
        for (int i = 0; i < 2; ++i) {
            const bf16_t* pr = &sA[b][wm + i * 16 + lm][hl * 8];
            fa[i].h[0] = *(const v8bf_t*)pr;
            fa[i].h[1] = *(const v8bf_t*)(pr + 16);
        }
        // B fragment (32x16 bf16): lanes 0-15 col=lm K{0-7,16-23};
        // lanes 16-31 col=lm K{8-15,24-31}
        #pragma unroll
        for (int t = 0; t < 4; ++t) {
            const bf16_t* pc = &sB[b][wn + t * 16 + lm][hl * 8];
            fb[t].h[0] = *(const v8bf_t*)pc;
            fb[t].h[1] = *(const v8bf_t*)(pc + 16);
        }
        #pragma unroll
        for (int i = 0; i < 2; ++i)
            #pragma unroll
            for (int t = 0; t < 4; ++t)
                acc[i][t] = __builtin_amdgcn_wmma_f32_16x16x32_bf16(
                    false, fa[i].v, false, fb[t].v,
                    (short)0, acc[i][t], false, false);
    }

    // epilogue: C/D layout -> lane holds column lm, rows hl*8 + e
    const int crow = row0 + wm + hl * 8;
    #pragma unroll
    for (int t = 0; t < 4; ++t) {
        int col  = col0 + wn + t * 16 + lm;
        float bv = bias ? bias[col] : 0.f;
        #pragma unroll
        for (int i = 0; i < 2; ++i) {
            #pragma unroll
            for (int e = 0; e < 8; ++e) {
                size_t idx = (size_t)(crow + i * 16 + e) * N + col;
                float val  = acc[i][t][e] + bv;
                if constexpr (EPI == EPI_F32) {
                    ((float*)Cv)[idx] = val;
                } else if constexpr (EPI == EPI_BF16) {
                    ((bf16_t*)Cv)[idx] = (bf16_t)val;
                } else if constexpr (EPI == EPI_GELU_BF16) {
                    float g = 0.5f * val * (1.f + erff(val * 0.70710678f));
                    ((bf16_t*)Cv)[idx] = (bf16_t)g;
                } else { // EPI_ADD_F32
                    float* C = (float*)Cv;
                    C[idx] = C[idx] + val;
                }
            }
        }
    }
}

// =====================================================================
// launcher
// =====================================================================
extern "C" void kernel_launch(void* const* d_in, const int* in_sizes, int n_in,
                              void* d_out, int out_size, void* d_ws, size_t ws_size,
                              hipStream_t stream)
{
    const float* x   = (const float*)d_in[0];
    const float* n1s = (const float*)d_in[1];
    const float* n1b = (const float*)d_in[2];
    const float* n2s = (const float*)d_in[3];
    const float* n2b = (const float*)d_in[4];
    const float* q_w = (const float*)d_in[5];
    const float* tw1 = (const float*)d_in[6] + 4 * (size_t)kD * kHid;  // titan_w1[4]
    const float* tb1 = (const float*)d_in[7] + 4 * (size_t)kHid;       // titan_b1[4]
    const float* tw2 = (const float*)d_in[8] + 4 * (size_t)kHid * kD;  // titan_w2[4]
    const float* tb2 = (const float*)d_in[9] + 4 * (size_t)kD;         // titan_b2[4]
    const float* ow  = (const float*)d_in[10];
    const float* ob  = (const float*)d_in[11];
    const float* cw1 = (const float*)d_in[12];
    const float* cb1 = (const float*)d_in[13];
    const float* cw2 = (const float*)d_in[14];
    const float* cb2 = (const float*)d_in[15];
    float* outp = (float*)d_out;

    // workspace carve-out
    size_t off = 0;
    auto carve = [&](size_t bytes) -> char* {
        char* p = (char*)d_ws + off;
        off += (bytes + 255) & ~(size_t)255;
        return p;
    };
    bf16_t* ybf = (bf16_t*)carve((size_t)kM * kD * 2);    // LN1 out, later LN2 out
    float*  qf  = (float*) carve((size_t)kM * kD * 4);    // q fp32, later out-proj fp32
    bf16_t* qbf = (bf16_t*)carve((size_t)kM * kD * 2);    // q after rope
    bf16_t* hbf = (bf16_t*)carve((size_t)kM * kHid * 2);  // hidden, reused per MLP
    bf16_t* ubf = (bf16_t*)carve((size_t)kM * kD * 2);    // titan MLP output
    // all weights stored TRANSPOSED (N x K) in bf16
    bf16_t* qwb = (bf16_t*)carve((size_t)kD * kD * 2);
    bf16_t* t1b = (bf16_t*)carve((size_t)kD * kHid * 2);
    bf16_t* t2b = (bf16_t*)carve((size_t)kHid * kD * 2);
    bf16_t* owb = (bf16_t*)carve((size_t)kD * kD * 2);
    bf16_t* c1b = (bf16_t*)carve(3 * (size_t)kD * kHid * 2);
    bf16_t* c2b = (bf16_t*)carve(3 * (size_t)kHid * kD * 2);

    auto cvt = [&](const float* src, bf16_t* dst, int K, int N) {
        int n = K * N;
        cvt_tr_kernel<<<(n + 255) / 256, 256, 0, stream>>>(src, dst, K, N);
    };
    cvt(q_w, qwb, kD, kD);
    cvt(tw1, t1b, kD, kHid);
    cvt(tw2, t2b, kHid, kD);
    cvt(ow,  owb, kD, kD);
    for (int i = 0; i < 3; ++i) {
        cvt(cw1 + (size_t)i * kD * kHid, c1b + (size_t)i * kD * kHid, kD, kHid);
        cvt(cw2 + (size_t)i * kHid * kD, c2b + (size_t)i * kHid * kD, kHid, kD);
    }

    dim3 blk(256);
    // y = LN1(x)
    resid_ln_kernel<<<kM / 8, blk, 0, stream>>>(x, nullptr, n1s, n1b, nullptr, ybf);
    // q = y @ q_w
    gemm_bf16_kernel<EPI_F32><<<dim3(kM / 128, kD / 128), blk, 0, stream>>>(
        ybf, qwb, nullptr, qf, kD, kD);
    // q = rope3d(q)
    rope_kernel<<<(kM * 192) / 256, blk, 0, stream>>>(qf, qbf);
    // titan MLP
    gemm_bf16_kernel<EPI_GELU_BF16><<<dim3(kM / 128, kHid / 128), blk, 0, stream>>>(
        qbf, t1b, tb1, hbf, kD, kHid);
    gemm_bf16_kernel<EPI_BF16><<<dim3(kM / 128, kD / 128), blk, 0, stream>>>(
        hbf, t2b, tb2, ubf, kHid, kD);
    // out projection -> qf (fp32)
    gemm_bf16_kernel<EPI_F32><<<dim3(kM / 128, kD / 128), blk, 0, stream>>>(
        ubf, owb, ob, qf, kD, kD);
    // x2 = x + out -> d_out ; y2 = LN2(x2) -> ybf
    resid_ln_kernel<<<kM / 8, blk, 0, stream>>>(x, qf, n2s, n2b, outp, ybf);
    // d_out += sum_i cms MLP_i(y2)
    for (int i = 0; i < 3; ++i) {
        gemm_bf16_kernel<EPI_GELU_BF16><<<dim3(kM / 128, kHid / 128), blk, 0, stream>>>(
            ybf, c1b + (size_t)i * kD * kHid, cb1 + (size_t)i * kHid, hbf, kD, kHid);
        gemm_bf16_kernel<EPI_ADD_F32><<<dim3(kM / 128, kD / 128), blk, 0, stream>>>(
            hbf, c2b + (size_t)i * kHid * kD, cb2 + (size_t)i * kD, outp, kHid, kD);
    }
    (void)in_sizes; (void)n_in; (void)out_size; (void)ws_size;
}
